// SkipGram_56289841381794
// MI455X (gfx1250) — matrix operand order repaired and checked by reading, not compiled, not run on hardware
//
#include <hip/hip_runtime.h>
#include <hip/hip_bf16.h>

// ---------------------------------------------------------------------------
// SkipGram forward: out = log_softmax((gather(W1, idx) + b1) @ W2^T + b2)
//   B=1024, D=128, V=100000.
//
// MI455X roofline: output write is 1024*100000*4B = 409.6MB -> ~17.6us floor
// @ 23.3TB/s HBM. W2 (51.2MB) fits in the 192MB L2, so recomputing logits in
// two passes (sum-exp pass + write pass) moves ~460MB total vs 1.23GB for a
// store/reload log-softmax. GEMM math (2 x 26.2 GFLOP, bf16 WMMA w/ f32 acc)
// is far below the matrix ceiling.
//
//   k0: zero per-row sum-exp accumulators (1024 f32 in ws)
//   k1: hidden[b][k] = bf16(W1[k][idx[b]] + b1[k])        (256KB in ws)
//   k2: pass A — per-row sum(exp(logit)) via atomicAdd (logits ~1e-2 scale,
//       exp is safe in f32 without max-subtraction; sum ~1e5 cannot overflow)
//   k3: pass B — recompute logits, write out = logit - log(sum)
//
// Each wave processes MB=4 M-blocks so every converted W2 B-fragment feeds
// 4 WMMAs instead of 1 (cvt:wmma drops 8:1 -> 2:1, L2 read traffic /4).
// ---------------------------------------------------------------------------

#define VOCAB   100000
#define DIM     128
#define BATCH   1024
#define NTILES  (VOCAB / 16)    // 6250
#define MBLOCKS (BATCH / 16)    // 64
#define MB      4               // M-blocks (16 rows each) per wave
#define MGROUPS (MBLOCKS / MB)  // 16
#define NCHUNKS 250             // waves per M-group; 6250/250 = 25 tiles each

typedef __attribute__((ext_vector_type(16))) __bf16 v16bf;
typedef __attribute__((ext_vector_type(16))) float  v16f;
typedef __attribute__((ext_vector_type(8)))  float  v8f;

// ---------------- k0: zero the 1024 row accumulators -----------------------
__global__ void zero_sums_kernel(float* __restrict__ sums) {
    int t = blockIdx.x * blockDim.x + threadIdx.x;
    if (t < BATCH) sums[t] = 0.0f;
}

// ---------------- k1: hidden gather + bias, cast to bf16 -------------------
__global__ void gather_hidden_kernel(const int* __restrict__ idx,
                                     const float* __restrict__ W1,
                                     const float* __restrict__ b1,
                                     __bf16* __restrict__ hid) {
    int t = blockIdx.x * blockDim.x + threadIdx.x;   // BATCH*DIM threads
    int b = t >> 7;
    int k = t & (DIM - 1);
    float v = W1[(size_t)k * VOCAB + idx[b]] + b1[k];
    hid[t] = (__bf16)v;
}

// ---- load one 16x128 A block as 4 bf16 16x16x32-WMMA A fragments ----------
// A layout used here (16-bit A 16x32, contiguous k order, same as B): lane L -> row L&15,
//   elems 0..15 = K base..base+15 contiguous, base=(L>>4)*16
__device__ __forceinline__ void load_a_frags(const __bf16* __restrict__ hid,
                                             int mbase, int lane, v16bf a[4]) {
    int row  = mbase + (lane & 15);
    int base = (lane >> 4) * 16;
    const __bf16* hr = hid + row * DIM;
#pragma unroll
    for (int kk = 0; kk < 4; ++kk) {
        uint4 lo = *(const uint4*)(hr + kk * 32 + base);        // 8 bf16
        uint4 hi = *(const uint4*)(hr + kk * 32 + base + 8);    // 8 bf16
        v16bf av;
        ((uint4*)&av)[0] = lo;
        ((uint4*)&av)[1] = hi;
        a[kk] = av;
    }
}

// ---- one 16-col tile for MB M-blocks: 4 B-frags, 4*MB WMMAs ---------------
// B layout (mirrors the sparse-B doc): lane L holds col nbase+(L&15),
//   K = (L>>4)*16 .. +15 contiguous within each K32 chunk -> one 64B f32 load
//   per chunk from the W2 row, converted to bf16 in regs (W2 stays f32 in L2);
//   each converted fragment is reused by all MB accumulators.
__device__ __forceinline__ void tile_logits_mb(const v16bf a[MB][4],
                                               const float* __restrict__ W2,
                                               int lane, int nbase, v8f c[MB]) {
    int col = nbase + (lane & 15);
    int kb  = (lane >> 4) * 16;
    const float* wr = W2 + (size_t)col * DIM + kb;
#pragma unroll
    for (int m = 0; m < MB; ++m)
        c[m] = (v8f){0.f, 0.f, 0.f, 0.f, 0.f, 0.f, 0.f, 0.f};
#pragma unroll
    for (int kk = 0; kk < 4; ++kk) {
        v16f  wv = *(const v16f*)(wr + kk * 32);          // 64B, 64B-aligned
        v16bf bv = __builtin_convertvector(wv, v16bf);    // v_cvt_pk_bf16_f32
#pragma unroll
        for (int m = 0; m < MB; ++m)                      // independent WMMAs
            c[m] = __builtin_amdgcn_wmma_f32_16x16x32_bf16(
                       /*neg_a=*/false, a[m][kk], /*neg_b=*/false, bv,
                       /*c_mod=*/(short)0, c[m], /*reuse=*/false, false);
    }
}

// ---------------- k2: pass A — per-row sum(exp(logits)) --------------------
// C layout: c[m][j] <-> row mg + m*16 + (lane>>4)*8 + j, col nbase+(lane&15).
__global__ void __launch_bounds__(32)
pass_sumexp_kernel(const __bf16* __restrict__ hid,
                   const float* __restrict__ W2,
                   const float* __restrict__ b2,
                   float* __restrict__ sums) {
    int lane = threadIdx.x;
    int mg   = blockIdx.x * (MB * 16);

    v16bf a[MB][4];
#pragma unroll
    for (int m = 0; m < MB; ++m)
        load_a_frags(hid, mg + m * 16, lane, a[m]);

    float acc[MB][8];
#pragma unroll
    for (int m = 0; m < MB; ++m)
#pragma unroll
        for (int j = 0; j < 8; ++j) acc[m][j] = 0.f;

    for (int t = blockIdx.y; t < NTILES; t += gridDim.y) {
        int nbase = t * 16;
        v8f c[MB];
        tile_logits_mb(a, W2, lane, nbase, c);
        float b2v = b2[nbase + (lane & 15)];
#pragma unroll
        for (int m = 0; m < MB; ++m)
#pragma unroll
            for (int j = 0; j < 8; ++j)
                acc[m][j] += __expf(c[m][j] + b2v);
    }
    // reduce each row across its 16 columns (lanes 0-15 / 16-31 hold
    // disjoint row groups, so a width-16 xor tree is exact)
#pragma unroll
    for (int m = 0; m < MB; ++m)
#pragma unroll
        for (int j = 0; j < 8; ++j) {
            acc[m][j] += __shfl_xor(acc[m][j], 8, 16);
            acc[m][j] += __shfl_xor(acc[m][j], 4, 16);
            acc[m][j] += __shfl_xor(acc[m][j], 2, 16);
            acc[m][j] += __shfl_xor(acc[m][j], 1, 16);
        }
    if ((lane & 15) == 0) {
        int half = lane >> 4;
#pragma unroll
        for (int m = 0; m < MB; ++m)
#pragma unroll
            for (int j = 0; j < 8; ++j)
                atomicAdd(&sums[mg + m * 16 + half * 8 + j], acc[m][j]);
    }
}

// ---------------- k3: pass B — recompute logits, write log-softmax ---------
__global__ void __launch_bounds__(32)
pass_write_kernel(const __bf16* __restrict__ hid,
                  const float* __restrict__ W2,
                  const float* __restrict__ b2,
                  const float* __restrict__ sums,
                  float* __restrict__ out) {
    int lane = threadIdx.x;
    int mg   = blockIdx.x * (MB * 16);
    int half = lane >> 4;

    v16bf a[MB][4];
#pragma unroll
    for (int m = 0; m < MB; ++m)
        load_a_frags(hid, mg + m * 16, lane, a[m]);

    float lse[MB][8];
#pragma unroll
    for (int m = 0; m < MB; ++m)
#pragma unroll
        for (int j = 0; j < 8; ++j)
            lse[m][j] = logf(sums[mg + m * 16 + half * 8 + j]);

    for (int t = blockIdx.y; t < NTILES; t += gridDim.y) {
        int nbase = t * 16;
        v8f c[MB];
        tile_logits_mb(a, W2, lane, nbase, c);
        int   col = nbase + (lane & 15);
        float b2v = b2[col];
#pragma unroll
        for (int m = 0; m < MB; ++m) {
            int rbase = mg + m * 16 + half * 8;
#pragma unroll
            for (int j = 0; j < 8; ++j)   // 16 lanes/row -> 64B coalesced
                out[(size_t)(rbase + j) * VOCAB + col] = c[m][j] + b2v - lse[m][j];
        }
    }
}

// ---------------------------------------------------------------------------
extern "C" void kernel_launch(void* const* d_in, const int* in_sizes, int n_in,
                              void* d_out, int out_size, void* d_ws, size_t ws_size,
                              hipStream_t stream) {
    const int*   idx = (const int*)d_in[0];
    const float* W1  = (const float*)d_in[1];
    const float* b1  = (const float*)d_in[2];
    const float* W2  = (const float*)d_in[3];
    const float* b2  = (const float*)d_in[4];
    float*       out = (float*)d_out;

    // ws layout: [0,4KB) row sum-exp accumulators; [4KB, 4KB+256KB) bf16 hidden
    float*  sums = (float*)d_ws;
    __bf16* hid  = (__bf16*)((char*)d_ws + 4096);

    zero_sums_kernel<<<4, 256, 0, stream>>>(sums);
    gather_hidden_kernel<<<(BATCH * DIM) / 256, 256, 0, stream>>>(idx, W1, b1, hid);

    dim3 gA(MGROUPS, NCHUNKS);   // 4000 waves, 25 tiles x 4 M-blocks each
    pass_sumexp_kernel<<<gA, 32, 0, stream>>>(hid, W2, b2, sums);

    dim3 gB(MGROUPS, NCHUNKS);   // store-bound pass: 410MB @ 23.3TB/s floor
    pass_write_kernel<<<gB, 32, 0, stream>>>(hid, W2, b2, sums, out);
}
